// VectorQuantizer_76501957476536
// MI455X (gfx1250) — matrix operand order repaired and checked by reading, not compiled, not run on hardware
//
#include <hip/hip_runtime.h>
#include <hip/hip_bf16.h>

typedef __attribute__((ext_vector_type(16))) __bf16 v16bf;
typedef __attribute__((ext_vector_type(8)))  float  v8f;

union ABReg { v16bf v; uint4 q[2]; };

#define VQ_N 16384
#define VQ_D 256
#define VQ_K 4096

// gfx1250 async global->LDS copies (ASYNCcnt path); fall back to register
// staging if the toolchain lacks the builtin.
#if defined(__AMDGCN__) && __has_builtin(__builtin_amdgcn_global_load_async_to_lds_b128)
#define VQ_ASYNC 1
// Builtin signature (from clang diagnostics): (int4 as(1)*, int4 as(3)*, imm, imm)
typedef __attribute__((__vector_size__(4 * sizeof(int)))) int vq_v4i;
typedef __attribute__((address_space(1))) vq_v4i vq_v4i_g;
typedef __attribute__((address_space(3))) vq_v4i vq_v4i_l;
#else
#define VQ_ASYNC 0
#endif

#if defined(__AMDGCN__) && __has_builtin(__builtin_amdgcn_s_wait_asynccnt)
#define VQ_WAIT_ASYNC() __builtin_amdgcn_s_wait_asynccnt(0)
#elif defined(__AMDGCN__)
#define VQ_WAIT_ASYNC() asm volatile("s_wait_asynccnt 0x0" ::: "memory")
#else
#define VQ_WAIT_ASYNC()
#endif

// LDS B-tile geometry: rows padded to 264 bf16 (528B = 33*16B, and 132 dwords
// == 4 mod 64 banks -> 16 lanes of a half-wave hit distinct bank groups).
#define LROW   264
#define LTILE  (16 * LROW)          // one 16-code tile (elements)
// smem slot: [buf(2)][tile(2)][mat hi/lo(2)] of LTILE
#define LOFF(buf, tile, mat) ((((buf) * 2 + (tile)) * 2 + (mat)) * LTILE)

// ---- bf16 helpers (bit-level, round-to-nearest-even) ----
__device__ __forceinline__ unsigned short f2bf(float x) {
    unsigned int u = __float_as_uint(x);
    u += 0x7FFFu + ((u >> 16) & 1u);
    return (unsigned short)(u >> 16);
}
__device__ __forceinline__ float bf2f(unsigned short h) {
    return __uint_as_float(((unsigned int)h) << 16);
}

// ---------------------------------------------------------------------------
// Kernel 1: pack z (B,D,H,W) fp32 -> z_flat (N,D) as bf16 hi/lo, LDS-tiled
// transpose so both the global read (along hw) and write (along d) coalesce.
// ---------------------------------------------------------------------------
__global__ __launch_bounds__(256) void vq_pack_z(const float* __restrict__ z,
                                                 unsigned short* __restrict__ zhi,
                                                 unsigned short* __restrict__ zlo) {
    __shared__ float tile[32][33];
    const int b   = blockIdx.z;
    const int d0  = blockIdx.x * 32;
    const int hw0 = blockIdx.y * 32;
    const int tx  = threadIdx.x;   // 0..31
    const int ty  = threadIdx.y;   // 0..7
    #pragma unroll
    for (int i = ty; i < 32; i += 8) {
        tile[i][tx] = z[(((b * 256) + d0 + i) << 10) + hw0 + tx];
    }
    __syncthreads();
    #pragma unroll
    for (int i = ty; i < 32; i += 8) {
        const int n = (b << 10) + hw0 + i;      // n = b*1024 + hw
        const float x = tile[tx][i];            // d = d0+tx
        const unsigned short h = f2bf(x);
        const unsigned short l = f2bf(x - bf2f(h));
        zhi[n * VQ_D + d0 + tx] = h;
        zlo[n * VQ_D + d0 + tx] = l;
    }
}

// ---------------------------------------------------------------------------
// Kernel 2: pack emb (K,D) -> bf16 hi/lo + row norms e2[k]; also zero hist.
// ---------------------------------------------------------------------------
__global__ __launch_bounds__(256) void vq_pack_emb(const float* __restrict__ emb,
                                                   unsigned short* __restrict__ ehi,
                                                   unsigned short* __restrict__ elo,
                                                   float* __restrict__ e2,
                                                   int* __restrict__ hist) {
    __shared__ float red[256];
    const int k = blockIdx.x;
    const int d = threadIdx.x;
    const float e = emb[k * VQ_D + d];
    const unsigned short h = f2bf(e);
    const unsigned short l = f2bf(e - bf2f(h));
    ehi[k * VQ_D + d] = h;
    elo[k * VQ_D + d] = l;
    red[d] = e * e;
    __syncthreads();
    for (int s = 128; s > 0; s >>= 1) {
        if (d < s) red[d] += red[d + s];
        __syncthreads();
    }
    if (d == 0) { e2[k] = red[0]; hist[k] = 0; }
}

// ---------------------------------------------------------------------------
// Kernel 3: WMMA distance GEMM + argmin.
//   - 8 waves/block; each wave owns a 16-row A stripe held in registers
//     (bf16 hi+lo, ISA A-layout, 128 VGPRs).
//   - Codebook B staged through double-buffered LDS (2 k-tiles = 32 codes per
//     stage, hi+lo), shared by all 8 waves -> 8x cut in L2 traffic.
//   - Stage s+1 is fetched with GLOBAL_LOAD_ASYNC_TO_LDS_B128 (ASYNCcnt):
//     no VGPR staging (no spills), copies overlap the 48 WMMAs of stage s,
//     one s_wait_asynccnt + one barrier per stage.
//   - Single accumulator chain per tile (D->C accumulation chaining is the
//     normal XDL mode); two interleaved tiles give two independent chains.
//   - dot = ahi*bhi + ahi*blo + alo*bhi (bf16x3 ~= fp32 precision).
//   - score = e2[k] - 2*dot  (||z||^2 constant per row: dropped for argmin).
// ---------------------------------------------------------------------------
__global__ __launch_bounds__(256) void vq_argmin(const unsigned short* __restrict__ zhi,
                                                 const unsigned short* __restrict__ zlo,
                                                 const unsigned short* __restrict__ ehi,
                                                 const unsigned short* __restrict__ elo,
                                                 const float* __restrict__ e2,
                                                 int* __restrict__ idx_out) {
    __shared__ unsigned short smem[2 * 2 * 2 * LTILE];   // ~66 KB

    const int tid  = threadIdx.x;
    const int lane = tid & 31;
    const int wave = tid >> 5;
    const int n0   = blockIdx.x * 128 + wave * 16;
    const int half = lane >> 4;       // 0: K-halves {0..7,16..23}, 1: {8..15,24..31}
    const int l16  = lane & 15;
    const int row  = n0 + l16;

    // Staging assignment: 256 threads copy 32 codebook rows (8 threads/row,
    // 64B per thread per matrix).
    const int srow = tid >> 3;        // 0..31
    const int scol = (tid & 7) * 32;  // element offset within a 256-elem row
    const int ldh  = LOFF(0, srow >> 4, 0) + (srow & 15) * LROW + scol;  // buf-0 hi dst
    const int ldl  = LOFF(0, srow >> 4, 1) + (srow & 15) * LROW + scol;  // buf-0 lo dst
    const int lbuf = LOFF(1, 0, 0);   // element offset between buffers

    // Preload A: 8 chunks of K=32, hi and lo, per ISA 16-bit A layout.
    ABReg a_hi[8], a_lo[8];
    const unsigned short* zr_hi = zhi + (size_t)row * VQ_D;
    const unsigned short* zr_lo = zlo + (size_t)row * VQ_D;
    #pragma unroll
    for (int c = 0; c < 8; ++c) {
        const int off0 = c * 32 + half * 8;
        a_hi[c].q[0] = *(const uint4*)(zr_hi + off0);
        a_hi[c].q[1] = *(const uint4*)(zr_hi + off0 + 16);
        a_lo[c].q[0] = *(const uint4*)(zr_lo + off0);
        a_lo[c].q[1] = *(const uint4*)(zr_lo + off0 + 16);
    }

    float best[8];
    int   bidx[8];
    #pragma unroll
    for (int j = 0; j < 8; ++j) { best[j] = 3.4e38f; bidx[j] = 0; }

#if VQ_ASYNC
    // ---- async stage issue: copy stage `st` (32 codes) into buffer `buf` ----
    #define VQ_STAGE_ASYNC(st, buf)                                                        \
        do {                                                                               \
            const unsigned short* gh = ehi + ((size_t)(st) * 32 + srow) * VQ_D + scol;     \
            const unsigned short* gl = elo + ((size_t)(st) * 32 + srow) * VQ_D + scol;     \
            unsigned short* dh = smem + (buf) * lbuf + ldh;                                \
            unsigned short* dl = smem + (buf) * lbuf + ldl;                                \
            _Pragma("unroll")                                                              \
            for (int q = 0; q < 4; ++q) {                                                  \
                __builtin_amdgcn_global_load_async_to_lds_b128(                            \
                    (vq_v4i_g*)(gh + q * 8), (vq_v4i_l*)(dh + q * 8), 0, 0);               \
                __builtin_amdgcn_global_load_async_to_lds_b128(                            \
                    (vq_v4i_g*)(gl + q * 8), (vq_v4i_l*)(dl + q * 8), 0, 0);               \
            }                                                                              \
        } while (0)

    VQ_STAGE_ASYNC(0, 0);
    VQ_WAIT_ASYNC();
    __syncthreads();
#else
    uint4 ph[4], pl[4];
    {
        const unsigned short* gh = ehi + (size_t)srow * VQ_D + scol;
        const unsigned short* gl = elo + (size_t)srow * VQ_D + scol;
        #pragma unroll
        for (int q = 0; q < 4; ++q) {
            ph[q] = *(const uint4*)(gh + q * 8);
            pl[q] = *(const uint4*)(gl + q * 8);
        }
        #pragma unroll
        for (int q = 0; q < 4; ++q) {
            *(uint4*)(smem + ldh + q * 8) = ph[q];
            *(uint4*)(smem + ldl + q * 8) = pl[q];
        }
    }
    __syncthreads();
#endif

    for (int s = 0; s < VQ_K / 32; ++s) {
        const int cur = s & 1;

#if VQ_ASYNC
        // Issue next stage's async copies; they overlap the WMMAs below.
        // Buffer cur^1's readers finished before the previous barrier.
        if (s + 1 < VQ_K / 32) VQ_STAGE_ASYNC(s + 1, cur ^ 1);
#else
        if (s + 1 < VQ_K / 32) {
            const unsigned short* gh = ehi + ((size_t)(s + 1) * 32 + srow) * VQ_D + scol;
            const unsigned short* gl = elo + ((size_t)(s + 1) * 32 + srow) * VQ_D + scol;
            #pragma unroll
            for (int q = 0; q < 4; ++q) {
                ph[q] = *(const uint4*)(gh + q * 8);
                pl[q] = *(const uint4*)(gl + q * 8);
            }
        }
#endif

        // Compute the two k-tiles resident in buffer `cur`.
        #pragma unroll
        for (int tt = 0; tt < 2; ++tt) {
            const int code = (s * 2 + tt) * 16 + l16;   // B: lane%16 = column = code
            const unsigned short* er_hi = smem + LOFF(cur, tt, 0) + l16 * LROW + half * 8;
            const unsigned short* er_lo = smem + LOFF(cur, tt, 1) + l16 * LROW + half * 8;
            const float ek = e2[code];
            v8f acc = {};
            #pragma unroll
            for (int c = 0; c < 8; ++c) {
                ABReg b_hi, b_lo;
                b_hi.q[0] = *(const uint4*)(er_hi + c * 32);
                b_hi.q[1] = *(const uint4*)(er_hi + c * 32 + 16);
                b_lo.q[0] = *(const uint4*)(er_lo + c * 32);
                b_lo.q[1] = *(const uint4*)(er_lo + c * 32 + 16);
                acc = __builtin_amdgcn_wmma_f32_16x16x32_bf16(false, a_hi[c].v, false, b_hi.v, (short)0, acc, false, false);
                acc = __builtin_amdgcn_wmma_f32_16x16x32_bf16(false, a_hi[c].v, false, b_lo.v, (short)0, acc, false, false);
                acc = __builtin_amdgcn_wmma_f32_16x16x32_bf16(false, a_lo[c].v, false, b_hi.v, (short)0, acc, false, false);
            }
            #pragma unroll
            for (int j = 0; j < 8; ++j) {
                const float score = ek - 2.0f * acc[j];
                if (score < best[j]) { best[j] = score; bidx[j] = code; }
            }
        }

#if VQ_ASYNC
        // Own async copies done -> barrier publishes them (and retires all
        // reads of buffer `cur`) for the next iteration.
        VQ_WAIT_ASYNC();
        __syncthreads();
#else
        __syncthreads();
        if (s + 1 < VQ_K / 32) {
            unsigned short* dh = smem + (cur ^ 1) * lbuf + ldh;
            unsigned short* dl = smem + (cur ^ 1) * lbuf + ldl;
            #pragma unroll
            for (int q = 0; q < 4; ++q) {
                *(uint4*)(dh + q * 8) = ph[q];
                *(uint4*)(dl + q * 8) = pl[q];
            }
        }
        __syncthreads();
#endif
    }

    // C layout: lane l, slot j -> row (j + 8*half), col l%16. Min across the
    // 16 columns (lanes) per half; tie-break on lower code index (first-hit).
    #pragma unroll
    for (int j = 0; j < 8; ++j) {
        float b = best[j];
        int   bi = bidx[j];
        #pragma unroll
        for (int off = 8; off >= 1; off >>= 1) {
            const float ob = __shfl_xor(b, off, 16);
            const int   oi = __shfl_xor(bi, off, 16);
            if (ob < b || (ob == b && oi < bi)) { b = ob; bi = oi; }
        }
        if (l16 == 0) idx_out[n0 + j + half * 8] = bi;
    }
}

// ---------------------------------------------------------------------------
// Kernel 4: quantized output (= emb gather, plain reshape -> row-major N,D)
// + deterministic per-block partial sums of (q - z)^2.
// ---------------------------------------------------------------------------
__global__ __launch_bounds__(256) void vq_quant_loss(const float* __restrict__ z,
                                                     const float* __restrict__ emb,
                                                     const int* __restrict__ idx,
                                                     float* __restrict__ q_out,
                                                     float* __restrict__ partials) {
    __shared__ float red[256];
    const int tid = blockIdx.x * 256 + threadIdx.x;   // [0, N*D)
    const int n = tid >> 8;
    const int d = tid & 255;
    const int b = n >> 10;
    const int hw = n & 1023;
    const float zv = z[(((b * 256) + d) << 10) + hw];
    const float qv = emb[(size_t)idx[n] * VQ_D + d];
    q_out[tid] = qv;
    const float diff = qv - zv;
    red[threadIdx.x] = diff * diff;
    __syncthreads();
    for (int s = 128; s > 0; s >>= 1) {
        if (threadIdx.x < s) red[threadIdx.x] += red[threadIdx.x + s];
        __syncthreads();
    }
    if (threadIdx.x == 0) partials[blockIdx.x] = red[0];
}

// ---------------------------------------------------------------------------
// Kernel 5: histogram of indices (integer atomics: deterministic).
// ---------------------------------------------------------------------------
__global__ __launch_bounds__(256) void vq_hist(const int* __restrict__ idx,
                                               int* __restrict__ hist) {
    const int n = blockIdx.x * 256 + threadIdx.x;
    atomicAdd(&hist[idx[n]], 1);
}

// ---------------------------------------------------------------------------
// Kernel 6: one-hot encodings, 256 MB streaming write (the bandwidth hog).
// float2 stores: encodings segment of d_out is 8B- but not 16B-aligned.
// ---------------------------------------------------------------------------
__global__ __launch_bounds__(256) void vq_encodings(const int* __restrict__ idx,
                                                    float* __restrict__ enc) {
    const unsigned int t = blockIdx.x * 256u + threadIdx.x;
    const unsigned int base = t * 2u;
    const int n = (int)(base >> 12);
    const int k = (int)(base & 4095u);
    const int tgt = idx[n];
    float2 v;
    v.x = (k     == tgt) ? 1.0f : 0.0f;
    v.y = (k + 1 == tgt) ? 1.0f : 0.0f;
    *(float2*)(enc + base) = v;
}

// ---------------------------------------------------------------------------
// Kernel 7: final fixed-order reductions -> loss and perplexity scalars.
// ---------------------------------------------------------------------------
__global__ __launch_bounds__(256) void vq_finalize(const float* __restrict__ partials,
                                                   const int* __restrict__ hist,
                                                   float* __restrict__ out_scalars) {
    __shared__ float red[256];
    const int t = threadIdx.x;
    float s = 0.0f;
    for (int i = t; i < VQ_N * VQ_D / 256; i += 256) s += partials[i];
    red[t] = s;
    __syncthreads();
    for (int w = 128; w > 0; w >>= 1) {
        if (t < w) red[t] += red[t + w];
        __syncthreads();
    }
    const float lossv = 1.25f * red[0] / (float)(VQ_N * VQ_D);
    __syncthreads();
    float p = 0.0f;
    for (int i = t; i < VQ_K; i += 256) {
        const float pr = (float)hist[i] * (1.0f / (float)VQ_N);
        p += pr * logf(pr + 1e-10f);
    }
    red[t] = p;
    __syncthreads();
    for (int w = 128; w > 0; w >>= 1) {
        if (t < w) red[t] += red[t + w];
        __syncthreads();
    }
    if (t == 0) {
        out_scalars[0] = lossv;          // loss
        out_scalars[1] = expf(-red[0]);  // perplexity
    }
}

extern "C" void kernel_launch(void* const* d_in, const int* in_sizes, int n_in,
                              void* d_out, int out_size, void* d_ws, size_t ws_size,
                              hipStream_t stream) {
    const float* z   = (const float*)d_in[0];   // (16,256,32,32) fp32
    const float* emb = (const float*)d_in[1];   // (4096,256) fp32

    float* out = (float*)d_out;
    float* q_out       = out;                        // N*D = 4,194,304
    float* out_scalars = out + (size_t)VQ_N * VQ_D;  // [loss, perplexity]
    float* enc         = out + (size_t)VQ_N * VQ_D + 2;  // N*K one-hot

    char* ws = (char*)d_ws;
    unsigned short* zhi = (unsigned short*)(ws);
    unsigned short* zlo = (unsigned short*)(ws + 8u * 1024 * 1024);
    unsigned short* ehi = (unsigned short*)(ws + 16u * 1024 * 1024);
    unsigned short* elo = (unsigned short*)(ws + 18u * 1024 * 1024);
    float* e2           = (float*)(ws + 20u * 1024 * 1024);
    int*   idx          = (int*)  (ws + 20u * 1024 * 1024 + 64u * 1024);
    int*   hist         = (int*)  (ws + 20u * 1024 * 1024 + 128u * 1024);
    float* partials     = (float*)(ws + 20u * 1024 * 1024 + 192u * 1024);

    // 1) pack z -> bf16 hi/lo in (N,D) layout
    vq_pack_z<<<dim3(VQ_D / 32, 1024 / 32, 16), dim3(32, 8, 1), 0, stream>>>(z, zhi, zlo);
    // 2) pack emb + norms + zero hist
    vq_pack_emb<<<VQ_K, 256, 0, stream>>>(emb, ehi, elo, e2, hist);
    // 3) WMMA distance GEMM + argmin (8 waves/block share LDS-staged B)
    vq_argmin<<<VQ_N / 128, 256, 0, stream>>>(zhi, zlo, ehi, elo, e2, idx);
    // 4) quantized gather + loss partials
    vq_quant_loss<<<VQ_N * VQ_D / 256, 256, 0, stream>>>(z, emb, idx, q_out, partials);
    // 5) index histogram
    vq_hist<<<VQ_N / 256, 256, 0, stream>>>(idx, hist);
    // 6) one-hot encodings (bandwidth-dominant 256 MB write)
    vq_encodings<<<(VQ_N * (VQ_K / 2)) / 256, 256, 0, stream>>>(idx, enc);
    // 7) scalars
    vq_finalize<<<1, 256, 0, stream>>>(partials, hist, out_scalars);
}
